// VectorQuantizer_67912022885005
// MI455X (gfx1250) — compile-verified, matrix operand order checked by reading, not run on hardware
//
#include <hip/hip_runtime.h>

// Problem constants (B=32, T=2048, C=256, N=4096)
constexpr int kC     = 256;       // feature dim
constexpr int kN     = 4096;      // codebook entries
constexpr int kRows  = 65536;     // B*T
constexpr long kElem = (long)kRows * kC;

typedef __attribute__((ext_vector_type(16))) __bf16 v16bf;
typedef __attribute__((ext_vector_type(8)))  float  v8f;

union AFrag { v16bf v; unsigned short u[16]; };
union BFrag { v16bf v; uint4 q[2]; };

__device__ __forceinline__ unsigned short f2bf_rne(float f) {
  unsigned u = __float_as_uint(f);
  u += 0x7FFFu + ((u >> 16) & 1u);
  return (unsigned short)(u >> 16);
}
__device__ __forceinline__ float bf2f(unsigned short h) {
  return __uint_as_float(((unsigned)h) << 16);
}

// ---------------------------------------------------------------- init
__global__ void vq_init(int* __restrict__ counts) {
  int i = blockIdx.x * blockDim.x + threadIdx.x;
  if (i < kN) counts[i] = 0;
}

// ------------------------------------------------- weight prep: bf16 split + ||e||^2
__global__ void vq_prep_weight(const float* __restrict__ w,
                               unsigned short* __restrict__ wHi,
                               unsigned short* __restrict__ wLo,
                               float* __restrict__ embed2) {
  __shared__ float red[256];
  const int row = blockIdx.x;
  const int c   = threadIdx.x;
  float v = w[(size_t)row * kC + c];
  unsigned short hb = f2bf_rne(v);
  wHi[(size_t)row * kC + c] = hb;
  wLo[(size_t)row * kC + c] = f2bf_rne(v - bf2f(hb));
  red[c] = v * v;
  __syncthreads();
  for (int s = 128; s > 0; s >>= 1) {
    if (c < s) red[c] += red[c + s];
    __syncthreads();
  }
  if (c == 0) embed2[row] = red[0];
}

// ------------------------------------------------- fused GEMM + argmin
// 256 threads = 8 waves; each wave owns a 16-row tile; block covers 128 rows.
// B tiles (32 codebook rows, hi+lo) staged in LDS via async global->LDS DMA,
// double-buffered; all 8 waves consume the shared tile (8x less L2 traffic).
constexpr int kRowB = 528;            // padded LDS row stride in bytes (264 halves)
constexpr int kMatB = 32 * kRowB;     // one matrix (hi or lo) per pair: 16896 B
constexpr int kBufB = 2 * kMatB;      // hi+lo per pair buffer: 33792 B
constexpr int kLdsB = 2 * kBufB;      // double buffer: 67584 B

__global__ __launch_bounds__(256) void vq_argmin(
    const float* __restrict__ x,
    const unsigned short* __restrict__ wHi,
    const unsigned short* __restrict__ wLo,
    const float* __restrict__ embed2,
    int* __restrict__ idxOut,
    float* __restrict__ idxOutF) {
  extern __shared__ unsigned short smem[];
  const int tid   = threadIdx.x;
  const int wave  = tid >> 5;
  const int lane  = tid & 31;
  const int l16   = lane & 15;
  const int kSel  = (lane >> 4) << 3;           // 0 or 8 (half-lane K select)
  const int mBase = blockIdx.x * 128 + wave * 16;
  const int row   = mBase + l16;

  const unsigned ldsBase = (unsigned)(size_t)(void*)smem;   // LDS byte address (low 32 bits)
  const char*    sbase   = (const char*)smem;

  // Build A fragments (16 rows x K=256) in registers: bf16 hi + lo split of fp32 x.
  AFrag aHi[8], aLo[8];
#pragma unroll
  for (int kc = 0; kc < 8; ++kc) {
    const float* xr = x + (size_t)row * kC + kc * 32 + kSel;
    float4 f0 = *(const float4*)(xr + 0);
    float4 f1 = *(const float4*)(xr + 4);
    float4 f2 = *(const float4*)(xr + 16);
    float4 f3 = *(const float4*)(xr + 20);
    float vals[16] = {f0.x, f0.y, f0.z, f0.w, f1.x, f1.y, f1.z, f1.w,
                      f2.x, f2.y, f2.z, f2.w, f3.x, f3.y, f3.z, f3.w};
#pragma unroll
    for (int j = 0; j < 16; ++j) {
      unsigned short hb = f2bf_rne(vals[j]);
      aHi[kc].u[j] = hb;
      aLo[kc].u[j] = f2bf_rne(vals[j] - bf2f(hb));
    }
  }

  // Async-fill one pair-buffer (32 codebook rows x 256 halves, hi+lo) into LDS.
  // 2048 x 16B chunks; thread does 8 chunks; global base is uniform per step j.
  auto fill = [&](int pair, int bufSel) {
    const size_t gOff = (size_t)pair * 32 * kC;        // halves
    const unsigned short* baseHi = wHi + gOff;
    const unsigned short* baseLo = wLo + gOff;
    const unsigned bufOff = ldsBase + (unsigned)(bufSel * kBufB);
#pragma unroll
    for (int j = 0; j < 8; ++j) {
      const int rid = 8 * (j & 3) + wave;              // row 0..31 in pair
      const unsigned voff  = (unsigned)(rid * 512 + lane * 16);        // global byte offset
      const unsigned laddr = bufOff + (unsigned)((j >= 4 ? kMatB : 0) + rid * kRowB + lane * 16);
      const unsigned short* sb = (j >= 4) ? baseLo : baseHi;
      asm volatile("global_load_async_to_lds_b128 %0, %1, %2"
                   :: "v"(laddr), "v"(voff), "s"(sb)
                   : "memory");
    }
  };

  float best[8];
  int   bidx[8];
#pragma unroll
  for (int r = 0; r < 8; ++r) { best[r] = 3.4e38f; bidx[r] = 0; }

  constexpr int nPairs = kN / 32;   // 128 iterations, 2 x 16-col tiles each
  fill(0, 0);
  for (int p = 0; p < nPairs; ++p) {
    const int buf = p & 1;
    if (p + 1 < nPairs) {
      fill(p + 1, buf ^ 1);
      asm volatile("s_wait_asynccnt 0x8" ::: "memory");  // current buffer's 8 ops done
    } else {
      asm volatile("s_wait_asynccnt 0x0" ::: "memory");
    }
    __syncthreads();   // all waves' DMA for this buffer complete

    const int n0 = p * 32 + l16;
    const int n1 = n0 + 16;
    v8f acc0 = {0.f, 0.f, 0.f, 0.f, 0.f, 0.f, 0.f, 0.f};
    v8f acc1 = {0.f, 0.f, 0.f, 0.f, 0.f, 0.f, 0.f, 0.f};
    const unsigned bo = (unsigned)(buf * kBufB) + (unsigned)(l16 * kRowB) + (unsigned)(kSel * 2);
#pragma unroll
    for (int kc = 0; kc < 8; ++kc) {
      const unsigned o = bo + (unsigned)(kc * 64);
      BFrag b0h, b0l, b1h, b1l;
      b0h.q[0] = *(const uint4*)(sbase + o);
      b0h.q[1] = *(const uint4*)(sbase + o + 32);
      b1h.q[0] = *(const uint4*)(sbase + o + 16 * kRowB);
      b1h.q[1] = *(const uint4*)(sbase + o + 16 * kRowB + 32);
      b0l.q[0] = *(const uint4*)(sbase + o + kMatB);
      b0l.q[1] = *(const uint4*)(sbase + o + kMatB + 32);
      b1l.q[0] = *(const uint4*)(sbase + o + kMatB + 16 * kRowB);
      b1l.q[1] = *(const uint4*)(sbase + o + kMatB + 16 * kRowB + 32);
      // dot = hi*hi + lo*hi + hi*lo  (~fp32-accurate), f32 accumulate
      acc0 = __builtin_amdgcn_wmma_f32_16x16x32_bf16(false, aHi[kc].v, false, b0h.v, (short)0, acc0, false, false);
      acc0 = __builtin_amdgcn_wmma_f32_16x16x32_bf16(false, aLo[kc].v, false, b0h.v, (short)0, acc0, false, false);
      acc0 = __builtin_amdgcn_wmma_f32_16x16x32_bf16(false, aHi[kc].v, false, b0l.v, (short)0, acc0, false, false);
      acc1 = __builtin_amdgcn_wmma_f32_16x16x32_bf16(false, aHi[kc].v, false, b1h.v, (short)0, acc1, false, false);
      acc1 = __builtin_amdgcn_wmma_f32_16x16x32_bf16(false, aLo[kc].v, false, b1h.v, (short)0, acc1, false, false);
      acc1 = __builtin_amdgcn_wmma_f32_16x16x32_bf16(false, aHi[kc].v, false, b1l.v, (short)0, acc1, false, false);
    }
    const float e0 = embed2[n0];
    const float e1 = embed2[n1];
#pragma unroll
    for (int r = 0; r < 8; ++r) {
      float s0 = __builtin_fmaf(-2.f, acc0[r], e0);    // ||e||^2 - 2 x.e
      if (s0 < best[r]) { best[r] = s0; bidx[r] = n0; }
      float s1 = __builtin_fmaf(-2.f, acc1[r], e1);
      if (s1 < best[r]) { best[r] = s1; bidx[r] = n1; }
    }
    __syncthreads();   // everyone done reading this buffer; safe to overwrite
  }

  // C/D layout: VGPR r holds row r (lanes 0-15) / row r+8 (lanes 16-31), col = lane%16.
#pragma unroll
  for (int r = 0; r < 8; ++r) {
    float b = best[r];
    int   i = bidx[r];
#pragma unroll
    for (int off = 8; off >= 1; off >>= 1) {
      float ob = __shfl_xor(b, off, 16);
      int   oi = __shfl_xor(i, off, 16);
      if (ob < b || (ob == b && oi < i)) { b = ob; i = oi; }
    }
    if (l16 == 0) {
      int outRow = mBase + ((lane >> 4) << 3) + r;
      idxOut[outRow]  = i;
      idxOutF[outRow] = (float)i;
    }
  }
}

// ------------------------------------------------- gather + MSE partials + histogram
__global__ void vq_gather(const float* __restrict__ x,
                          const float* __restrict__ w,
                          const int* __restrict__ idx,
                          float* __restrict__ qOut,
                          float* __restrict__ partials,
                          int* __restrict__ counts) {
  __shared__ float red[256];
  const int row = blockIdx.x;
  const int c   = threadIdx.x;
  const int id  = idx[row];
  float q = w[(size_t)id * kC + c];
  size_t off = (size_t)row * kC + c;
  qOut[off] = q;
  float d = x[off] - q;
  red[c] = d * d;
  __syncthreads();
  for (int s = 128; s > 0; s >>= 1) {
    if (c < s) red[c] += red[c + s];
    __syncthreads();
  }
  if (c == 0) {
    partials[row] = red[0];
    atomicAdd(&counts[id], 1);   // integer atomic -> deterministic
  }
}

// ------------------------------------------------- final scalars (fixed-order reduction)
__global__ void vq_final(const float* __restrict__ partials,
                         const int* __restrict__ counts,
                         float* __restrict__ outDiff,
                         float* __restrict__ outEnt) {
  __shared__ float red[256];
  const int t = threadIdx.x;
  float s = 0.f;
  for (int i = t; i < kRows; i += 256) s += partials[i];
  red[t] = s;
  __syncthreads();
  for (int st = 128; st > 0; st >>= 1) { if (t < st) red[t] += red[t + st]; __syncthreads(); }
  float diff = red[0] / (float)kElem;
  __syncthreads();
  float h = 0.f;
  for (int i = t; i < kN; i += 256) {
    float p = (float)counts[i] / (float)kRows;
    h += p * logf(p + 1e-10f);
  }
  red[t] = h;
  __syncthreads();
  for (int st = 128; st > 0; st >>= 1) { if (t < st) red[t] += red[t + st]; __syncthreads(); }
  if (t == 0) {
    *outDiff = diff;
    *outEnt  = expf(-red[0]);
  }
}

// ---------------------------------------------------------------- launch
extern "C" void kernel_launch(void* const* d_in, const int* in_sizes, int n_in,
                              void* d_out, int out_size, void* d_ws, size_t ws_size,
                              hipStream_t stream) {
  const float* x = (const float*)d_in[0];   // (32,2048,256) fp32
  const float* w = (const float*)d_in[1];   // (4096,256)   fp32
  float* out = (float*)d_out;

  // workspace layout (~4.6 MB)
  char* ws = (char*)d_ws;
  unsigned short* wHi    = (unsigned short*)(ws);                              // 2 MB
  unsigned short* wLo    = (unsigned short*)(ws + (size_t)2 * 1024 * 1024);    // 2 MB
  float*          embed2 = (float*)(ws + (size_t)4 * 1024 * 1024);             // 16 KB
  int*            idx    = (int*)(ws + (size_t)4 * 1024 * 1024 + 16 * 1024);   // 256 KB
  int*            counts = (int*)(ws + (size_t)4 * 1024 * 1024 + 272 * 1024);  // 16 KB
  float*          parts  = (float*)(ws + (size_t)4 * 1024 * 1024 + 288 * 1024);// 256 KB

  // output layout: idx(65536) | quanted(16777216) | diff(1) | entropy(1)
  float* outIdx  = out;
  float* outQ    = out + kRows;
  float* outDiff = out + kRows + kElem;
  float* outEnt  = outDiff + 1;

  vq_init<<<(kN + 255) / 256, 256, 0, stream>>>(counts);
  vq_prep_weight<<<kN, 256, 0, stream>>>(w, wHi, wLo, embed2);
  vq_argmin<<<kRows / 128, 256, kLdsB, stream>>>(x, wHi, wLo, embed2, idx, outIdx);
  vq_gather<<<kRows, 256, 0, stream>>>(x, w, idx, outQ, parts, counts);
  vq_final<<<1, 256, 0, stream>>>(parts, counts, outDiff, outEnt);
}
